// Model_69183333204203
// MI455X (gfx1250) — matrix operand order repaired and checked by reading, not compiled, not run on hardware
//
#include <hip/hip_runtime.h>
#include <stdint.h>

typedef __attribute__((ext_vector_type(16))) _Float16 v16h;
typedef __attribute__((ext_vector_type(8)))  float    v8f;
typedef __attribute__((ext_vector_type(2)))  _Float16 h2f;

#define Bv 8
#define Nv 150
#define Sv 22
#define Ev 299
#define Mv 32
#define ROWS (Bv*Nv)   // 1200
#define WPB 8          // waves (i-nodes) per block in pair kernel
#define MTS 36         // padded LDS row stride for m tile (conflict-free, float4-aligned)

// ---------------- quaternion helpers ----------------
__device__ __forceinline__ void qmul(const float a[4], const float b[4], float o[4]) {
    o[0] = a[0]*b[0] - a[1]*b[1] - a[2]*b[2] - a[3]*b[3];
    o[1] = a[0]*b[1] + a[1]*b[0] + a[2]*b[3] - a[3]*b[2];
    o[2] = a[0]*b[2] - a[1]*b[3] + a[2]*b[0] + a[3]*b[1];
    o[3] = a[0]*b[3] + a[1]*b[2] - a[2]*b[1] + a[3]*b[0];
}
__device__ __forceinline__ void qrot(const float q[4], const float v[3], float o[3]) {
    float w = q[0];
    float t0 = 2.f*(q[2]*v[2] - q[3]*v[1]);
    float t1 = 2.f*(q[3]*v[0] - q[1]*v[2]);
    float t2 = 2.f*(q[1]*v[1] - q[2]*v[0]);
    o[0] = v[0] + w*t0 + (q[2]*t2 - q[3]*t1);
    o[1] = v[1] + w*t1 + (q[3]*t0 - q[1]*t2);
    o[2] = v[2] + w*t2 + (q[1]*t1 - q[2]*t0);
}
__device__ __forceinline__ void a_from_f4(v16h& a, float4 f0, float4 f1, float4 f2, float4 f3) {
    a[0]=(_Float16)f0.x; a[1]=(_Float16)f0.y; a[2]=(_Float16)f0.z; a[3]=(_Float16)f0.w;
    a[4]=(_Float16)f1.x; a[5]=(_Float16)f1.y; a[6]=(_Float16)f1.z; a[7]=(_Float16)f1.w;
    a[8]=(_Float16)f2.x; a[9]=(_Float16)f2.y; a[10]=(_Float16)f2.z; a[11]=(_Float16)f2.w;
    a[12]=(_Float16)f3.x; a[13]=(_Float16)f3.y; a[14]=(_Float16)f3.z; a[15]=(_Float16)f3.w;
}

// ---------------- layer-1 h assembly: h = [q(4), x(3), feat(22), t/1000, 0...] ----
__global__ void k_init_h(const float* __restrict__ q, const float* __restrict__ x,
                         const float* __restrict__ f, const int* __restrict__ t,
                         float* __restrict__ h) {
    int r = blockIdx.x * blockDim.x + threadIdx.x;
    if (r >= ROWS) return;
    float ft = (float)t[0] / 1000.0f;
    float* hr = h + r*64;
    hr[0]=q[r*4+0]; hr[1]=q[r*4+1]; hr[2]=q[r*4+2]; hr[3]=q[r*4+3];
    hr[4]=x[r*3+0]; hr[5]=x[r*3+1]; hr[6]=x[r*3+2];
    #pragma unroll
    for (int c = 0; c < Sv; ++c) hr[7+c] = f[r*Sv+c];
    hr[29] = ft;
    #pragma unroll
    for (int c = 30; c < 64; ++c) hr[c] = 0.0f;     // zero-pad -> guard-free WMMA loads
}

// ---------------- zero-padded [h || msum] concat, stride 96 ----------------
__global__ void k_concat(const float* __restrict__ h, const float* __restrict__ msum,
                         float* __restrict__ hcat, int hd) {
    int idx = blockIdx.x * blockDim.x + threadIdx.x;
    if (idx >= ROWS*96) return;
    int r = idx / 96, c = idx % 96;
    int fi = hd + Mv;
    float v = 0.f;
    if (c < hd)      v = h[r*64 + c];
    else if (c < fi) v = msum[r*Mv + (c - hd)];
    hcat[idx] = v;
}

// ---------------- per-node precompute: AB[row,0:32]=h@Wm_hi+bm, [32:64]=h@Wm_hj
template<int HD>
__global__ void k_node_pre(const float* __restrict__ h, const float* __restrict__ wm,
                           const float* __restrict__ bm, float* __restrict__ AB) {
    constexpr int KTN = (HD + 31) / 32;
    constexpr int K32 = KTN * 32;
    __shared__ h2f wpk[(K32/2)*64];        // zero-padded, half2-packed over K pairs
    int tid = threadIdx.x;
    for (int idx = tid; idx < (K32/2)*64; idx += blockDim.x) {
        int kp = idx >> 6, c = idx & 63;
        int k0 = 2*kp, k1 = k0 + 1;
        float v0 = 0.f, v1 = 0.f;
        if (k0 < HD) v0 = (c < Mv) ? wm[k0*Mv + c] : wm[(HD + k0)*Mv + (c - Mv)];
        if (k1 < HD) v1 = (c < Mv) ? wm[k1*Mv + c] : wm[(HD + k1)*Mv + (c - Mv)];
        h2f w; w[0] = (_Float16)v0; w[1] = (_Float16)v1;
        wpk[idx] = w;
    }
    __syncthreads();

    int ct = tid >> 5, lane = tid & 31, half = lane >> 4, l16 = lane & 15;
    int row = blockIdx.x*16 + l16;
    int col = ct*16 + l16;
    const float4* hp = (const float4*)(h + row*64);
    v8f c = {};
    #pragma unroll
    for (int kt = 0; kt < KTN; ++kt) {
        int base0 = kt*32 + (half ? 16 : 0);
        v16h a;
        a_from_f4(a, hp[(base0>>2)], hp[(base0>>2)+1], hp[(base0>>2)+2], hp[(base0>>2)+3]);
        v16h b;
        int kp0 = kt*16 + (half ? 8 : 0);      // K-pair index base
        #pragma unroll
        for (int p = 0; p < 8; ++p) {
            h2f w = wpk[(kp0 + p)*64 + col];
            b[2*p] = w[0]; b[2*p+1] = w[1];
        }
        c = __builtin_amdgcn_wmma_f32_16x16x32_f16(false, a, false, b, (short)0, c, false, false);
    }
    float bias = (col < Mv) ? bm[col] : 0.0f;
    #pragma unroll
    for (int r = 0; r < 8; ++r) {
        int rr = blockIdx.x*16 + r + 8*half;
        AB[rr*64 + col] = c[r] + bias;
    }
}

// ---------------- pair kernel: one wave per (b,i); 16-j tiles via WMMA ------
__global__ void k_pair(const float* __restrict__ qin, const float* __restrict__ xin,
                       const uint8_t* __restrict__ mask,
                       const float* __restrict__ AB,
                       const float* __restrict__ wm, const float* __restrict__ wq,
                       const float* __restrict__ bq,
                       float* __restrict__ msum,
                       float* __restrict__ qout, int qstride,
                       float* __restrict__ xout, int xstride,
                       int hd) {
    __shared__ float relW[Ev*Mv];          // one-hot relposenc -> row lookup table
    __shared__ float wqsT[6*Mv];           // Wq transposed [c][n]
    __shared__ float bqs[8];
    __shared__ float wgs[9*Mv];            // geometry weight rows
    __shared__ float mks[160];             // mask as float, padded
    __shared__ float mt[WPB][16*MTS];      // per-wave m tile, padded stride
    int tid = threadIdx.x;
    int b = blockIdx.x;
    for (int idx = tid; idx < Ev*Mv; idx += blockDim.x)
        relW[idx] = wm[(2*hd)*Mv + idx];
    for (int idx = tid; idx < 6*Mv; idx += blockDim.x)
        wqsT[idx] = wq[(idx & 31)*6 + (idx >> 5)];
    for (int idx = tid; idx < 9*Mv; idx += blockDim.x)
        wgs[idx] = wm[(2*hd + Ev)*Mv + idx];
    if (tid < 6) bqs[tid] = bq[tid];
    for (int idx = tid; idx < 160; idx += blockDim.x)
        mks[idx] = (idx < Nv && mask[b*Nv + idx]) ? 1.f : 0.f;
    __syncthreads();

    int wave = tid >> 5, lane = tid & 31, half = lane >> 4, l16 = lane & 15;
    int i = blockIdx.y * WPB + wave;
    if (i >= Nv) return;                   // no further block-wide barriers
    int rowI = b*Nv + i;

    float4 qiv = *(const float4*)(qin + rowI*4);
    float qi[4] = {qiv.x, qiv.y, qiv.z, qiv.w};
    float xi[3] = {xin[rowI*3+0], xin[rowI*3+1], xin[rowI*3+2]};
    float maskI = mks[i];
    float baseA0 = AB[rowI*64 + l16];          // h_i part (+bm)
    float baseA1 = AB[rowI*64 + 16 + l16];

    v16h bg0, bg1;                             // geometry B-tiles (K padded 9->32)
    #pragma unroll
    for (int e = 0; e < 16; ++e) {
        float v0 = 0.f, v1 = 0.f;
        if (!half && e < 9) { v0 = wgs[e*Mv + l16]; v1 = wgs[e*Mv + 16 + l16]; }
        bg0[e] = (_Float16)v0; bg1[e] = (_Float16)v1;
    }

    float cnt = 0.f;                           // n_nodes = sum(mask)
    for (int n = lane; n < Nv; n += 32) cnt += mks[n];
    for (int off = 16; off >= 1; off >>= 1) cnt += __shfl_xor(cnt, off, 32);

    float msA0 = 0.f, msA1 = 0.f;
    float accS[4] = {0,0,0,0};
    float accX[3] = {0,0,0};
    float* mrow = &mt[wave][0];

    for (int jt = 0; jt < 10; ++jt) {
        int j  = jt*16 + l16;                  // this lane's pair
        int jc = (j < Nv) ? j : (Nv-1);        // clamp: loads unconditional
        float validJ = (j < Nv) ? 1.f : 0.f;
        int rowJ = b*Nv + jc;
        float4 qjv = *(const float4*)(qin + rowJ*4);
        float qj[4] = {qjv.x, qjv.y, qjv.z, qjv.w};
        float xj[3] = {xin[rowJ*3+0], xin[rowJ*3+1], xin[rowJ*3+2]};
        float mmflag = validJ * maskI * mks[jc] * ((j != i) ? 1.f : 0.f);

        float qjinv[4] = {qj[0], -qj[1], -qj[2], -qj[3]};
        float diff[3]  = {xi[0]-xj[0], xi[1]-xj[1], xi[2]-xj[2]};
        float lx[3]; qrot(qjinv, diff, lx);
        float tq[4], lq[4];
        qmul(qjinv, qi, tq); qmul(tq, qj, lq);
        float d2 = diff[0]*diff[0]+diff[1]*diff[1]+diff[2]*diff[2];
        float dt = fabsf(qi[0]*qj[0]+qi[1]*qj[1]+qi[2]*qj[2]+qi[3]*qj[3]);
        float feat[9] = {lx[0],lx[1],lx[2], lq[0],lq[1],lq[2],lq[3], d2, dt};

        // A-tile: rows = j; K = 9 geometry features (zeroing handled post-WMMA)
        v16h a = {};
        if (!half) { for (int e = 0; e < 9; ++e) a[e] = (_Float16)feat[e]; }

        // C-tiles: A_i + B_j + relpos row (bm folded in baseA); clamped rows fixed post-WMMA
        v8f c0, c1;
        float rv[8];
        #pragma unroll
        for (int r = 0; r < 8; ++r) {
            int jr  = jt*16 + r + 8*half;
            int jrc = (jr < Nv) ? jr : (Nv-1);
            int rowJr = b*Nv + jrc;
            int rel = (Nv-1) + i - jrc;
            c0[r] = baseA0 + AB[rowJr*64 + 32 + l16]      + relW[rel*Mv + l16];
            c1[r] = baseA1 + AB[rowJr*64 + 32 + 16 + l16] + relW[rel*Mv + 16 + l16];
            rv[r] = ((jr < Nv) && (jr != i)) ? (maskI * mks[jrc]) : 0.f;
        }
        v8f d0 = __builtin_amdgcn_wmma_f32_16x16x32_f16(false, a, false, bg0, (short)0, c0, false, false);
        v8f d1 = __builtin_amdgcn_wmma_f32_16x16x32_f16(false, a, false, bg1, (short)0, c1, false, false);

        // mask/diag/pad zeroing as selects on D rows; msum reduce + LDS stage
        #pragma unroll
        for (int r = 0; r < 8; ++r) {
            float e0 = d0[r] * rv[r];
            float e1 = d1[r] * rv[r];
            msA0 += e0; msA1 += e1;
            int M = r + 8*half;
            mrow[M*MTS + l16]      = e0;
            mrow[M*MTS + 16 + l16] = e1;
        }
        asm volatile("s_wait_dscnt 0" ::: "memory");   // wave-local LDS RAW fence

        // delta = (m @ Wq + bq) * mm : each half computes 3 of 6 comps, float4 LDS
        float del[3];
        {
            int cb = half*3;
            float a0 = bqs[cb], a1 = bqs[cb+1], a2 = bqs[cb+2];
            const float4* m4 = (const float4*)(mrow + l16*MTS);
            const float4* w0 = (const float4*)(wqsT + (cb+0)*Mv);
            const float4* w1 = (const float4*)(wqsT + (cb+1)*Mv);
            const float4* w2 = (const float4*)(wqsT + (cb+2)*Mv);
            #pragma unroll
            for (int n4 = 0; n4 < 8; ++n4) {
                float4 mv = m4[n4];
                float4 u0 = w0[n4], u1 = w1[n4], u2 = w2[n4];
                a0 += mv.x*u0.x + mv.y*u0.y + mv.z*u0.z + mv.w*u0.w;
                a1 += mv.x*u1.x + mv.y*u1.y + mv.z*u1.z + mv.w*u1.w;
                a2 += mv.x*u2.x + mv.y*u2.y + mv.z*u2.z + mv.w*u2.w;
            }
            del[0] = a0*mmflag; del[1] = a1*mmflag; del[2] = a2*mmflag;
        }
        float delO[3];
        #pragma unroll
        for (int c2 = 0; c2 < 3; ++c2) delO[c2] = __shfl_xor(del[c2], 16, 32);
        float d6[6];
        if (!half) { d6[0]=del[0]; d6[1]=del[1]; d6[2]=del[2]; d6[3]=delO[0]; d6[4]=delO[1]; d6[5]=delO[2]; }
        else       { d6[0]=delO[0];d6[1]=delO[1];d6[2]=delO[2];d6[3]=del[0]; d6[4]=del[1]; d6[5]=del[2]; }

        if (half) {    // position term (unmasked sum over valid j per reference)
            float lxd[3] = {lx[0]+d6[3], lx[1]+d6[4], lx[2]+d6[5]};
            float rot[3]; qrot(qj, lxd, rot);
            accX[0] += validJ*(rot[0]+xj[0]);
            accX[1] += validJ*(rot[1]+xj[1]);
            accX[2] += validJ*(rot[2]+xj[2]);
        } else {       // quaternion term
            float vq[4] = {0.f, d6[0], d6[1], d6[2]};
            float qm[4]; qmul(lq, vq, qm);
            accS[0] += validJ*(lq[0]+qm[0]); accS[1] += validJ*(lq[1]+qm[1]);
            accS[2] += validJ*(lq[2]+qm[2]); accS[3] += validJ*(lq[3]+qm[3]);
        }
    }

    msA0 += __shfl_xor(msA0, 16, 32);
    msA1 += __shfl_xor(msA1, 16, 32);
    if (!half) {
        msum[rowI*Mv + l16]      = msA0;
        msum[rowI*Mv + 16 + l16] = msA1;
    }
    for (int off = 1; off < 16; off <<= 1) {
        #pragma unroll
        for (int c2=0;c2<4;++c2) accS[c2] += __shfl_xor(accS[c2], off, 32);
        #pragma unroll
        for (int c2=0;c2<3;++c2) accX[c2] += __shfl_xor(accX[c2], off, 32);
    }
    if (lane == 0) {   // upd_q = q * normalize(s) * conj(q)
        float nrm = sqrtf(accS[0]*accS[0]+accS[1]*accS[1]+accS[2]*accS[2]+accS[3]*accS[3]);
        nrm = fmaxf(nrm, 1e-12f);
        float lu[4] = {accS[0]/nrm, accS[1]/nrm, accS[2]/nrm, accS[3]/nrm};
        float t1[4]; qmul(qi, lu, t1);
        float qc[4] = {qi[0], -qi[1], -qi[2], -qi[3]};
        float uq[4]; qmul(t1, qc, uq);
        float* o = qout + rowI*qstride;
        o[0]=uq[0]; o[1]=uq[1]; o[2]=uq[2]; o[3]=uq[3];
    }
    if (lane == 16) {  // upd_x = accX / (n_nodes - 1)
        float inv = 1.0f / (cnt - 1.0f);
        float* o = xout + rowI*xstride;
        o[0]=accX[0]*inv; o[1]=accX[1]*inv; o[2]=accX[2]*inv;
    }
}

// ---------------- node update: h_next = relu(hcat @ Wf + bf) ---------------
template<int FI>
__global__ void k_node(const float* __restrict__ hcat, const float* __restrict__ wf,
                       const float* __restrict__ bf, float* __restrict__ hout) {
    constexpr int KTN = (FI + 31) / 32;
    constexpr int K32 = KTN * 32;
    __shared__ h2f wpk[(K32/2)*64];
    int tid = threadIdx.x;
    for (int idx = tid; idx < (K32/2)*64; idx += blockDim.x) {
        int kp = idx >> 6, c = idx & 63;
        int k0 = 2*kp, k1 = k0 + 1;
        float v0 = (k0 < FI) ? wf[k0*64 + c] : 0.f;
        float v1 = (k1 < FI) ? wf[k1*64 + c] : 0.f;
        h2f w; w[0] = (_Float16)v0; w[1] = (_Float16)v1;
        wpk[idx] = w;
    }
    __syncthreads();

    int ct = tid >> 5, lane = tid & 31, half = lane >> 4, l16 = lane & 15;
    int row = blockIdx.x*16 + l16;
    int col = ct*16 + l16;
    const float4* hp = (const float4*)(hcat + row*96);
    v8f c = {};
    #pragma unroll
    for (int kt = 0; kt < KTN; ++kt) {
        int base0 = kt*32 + (half ? 16 : 0);
        v16h a;
        a_from_f4(a, hp[(base0>>2)], hp[(base0>>2)+1], hp[(base0>>2)+2], hp[(base0>>2)+3]);
        v16h b;
        int kp0 = kt*16 + (half ? 8 : 0);
        #pragma unroll
        for (int p = 0; p < 8; ++p) {
            h2f w = wpk[(kp0 + p)*64 + col];
            b[2*p] = w[0]; b[2*p+1] = w[1];
        }
        c = __builtin_amdgcn_wmma_f32_16x16x32_f16(false, a, false, b, (short)0, c, false, false);
    }
    float bias = bf[col];
    #pragma unroll
    for (int r = 0; r < 8; ++r) {
        int rr = blockIdx.x*16 + r + 8*half;
        hout[rr*64 + col] = fmaxf(c[r] + bias, 0.f);
    }
}

extern "C" void kernel_launch(void* const* d_in, const int* in_sizes, int n_in,
                              void* d_out, int out_size, void* d_ws, size_t ws_size,
                              hipStream_t stream) {
    const float*   quats = (const float*)d_in[0];
    const float*   trans = (const float*)d_in[1];
    const float*   feats = (const float*)d_in[2];
    const uint8_t* mask  = (const uint8_t*)d_in[3];
    const int*     tptr  = (const int*)d_in[4];
    const float *wm[4], *bm[4], *wf[4], *bf[4], *wq[4], *bq[4];
    for (int l = 0; l < 4; ++l) {
        int base = 5 + 6*l;
        wm[l] = (const float*)d_in[base];     bm[l] = (const float*)d_in[base+1];
        wf[l] = (const float*)d_in[base+2];   bf[l] = (const float*)d_in[base+3];
        wq[l] = (const float*)d_in[base+4];   bq[l] = (const float*)d_in[base+5];
    }
    float* ws   = (float*)d_ws;
    float* h0   = ws;                 // 1200*64
    float* h1   = h0   + ROWS*64;
    float* AB   = h1   + ROWS*64;     // 1200*64
    float* msum = AB   + ROWS*64;     // 1200*32
    float* hcat = msum + ROWS*Mv;     // 1200*96
    float* qA   = hcat + ROWS*96;     // 1200*4
    float* xA   = qA   + ROWS*4;      // 1200*3
    float* qB   = xA   + ROWS*3;
    float* xB   = qB   + ROWS*4;

    k_init_h<<<dim3((ROWS+127)/128), 128, 0, stream>>>(quats, trans, feats, tptr, h0);

    const float* hin[4]  = {h0, h1, h0, h1};
    float*       hout[3] = {h1, h0, h1};
    const float* qi_[4]  = {quats, qA, qB, qA};
    const float* xi_[4]  = {trans, xA, xB, xA};
    float*       qo_[4]  = {qA, qB, qA, (float*)d_out};
    float*       xo_[4]  = {xA, xB, xA, (float*)d_out + 4};
    int qs_[4] = {4,4,4,7}, xs_[4] = {3,3,3,7};
    int hd_[4] = {30,64,64,64};

    for (int l = 0; l < 4; ++l) {
        if (l == 0) k_node_pre<30><<<dim3(ROWS/16), 128, 0, stream>>>(hin[l], wm[l], bm[l], AB);
        else        k_node_pre<64><<<dim3(ROWS/16), 128, 0, stream>>>(hin[l], wm[l], bm[l], AB);

        k_pair<<<dim3(Bv, (Nv + WPB - 1)/WPB), WPB*32, 0, stream>>>(
            qi_[l], xi_[l], mask, AB, wm[l], wq[l], bq[l],
            msum, qo_[l], qs_[l], xo_[l], xs_[l], hd_[l]);

        if (l < 3) {
            k_concat<<<dim3((ROWS*96 + 255)/256), 256, 0, stream>>>(hin[l], msum, hcat, hd_[l]);
            if (l == 0) k_node<62><<<dim3(ROWS/16), 128, 0, stream>>>(hcat, wf[l], bf[l], hout[l]);
            else        k_node<96><<<dim3(ROWS/16), 128, 0, stream>>>(hcat, wf[l], bf[l], hout[l]);
        }
    }
    (void)in_sizes; (void)n_in; (void)out_size; (void)ws_size;
}
